// Seq2SeqGRU_27676769255797
// MI455X (gfx1250) — compile-verified, matrix operand order checked by reading, not run on hardware
//
#include <hip/hip_runtime.h>
#include <math.h>

#define Bsz 128
#define Tsz 100
#define Csz 66
#define Hsz 1536
#define KX0 96
#define EPSV 1e-5f

typedef __attribute__((ext_vector_type(16))) __bf16 v16bf;
typedef __attribute__((ext_vector_type(8)))  float  v8f;

struct U256 { uint4 lo, hi; };

// Load 16 bf16 elements laid out for WMMA A/B operand:
// per-lane chunk0 = elements [0..7], chunk1 = elements [16..23]
// (the hi/lo half-wave +8 element shift is folded into the base pointer).
__device__ __forceinline__ v16bf ldAB(const unsigned short* p) {
  U256 t;
  t.lo = *reinterpret_cast<const uint4*>(p);
  t.hi = *reinterpret_cast<const uint4*>(p + 16);
  return __builtin_bit_cast(v16bf, t);
}

__device__ __forceinline__ unsigned short f2bf(float f) {
  unsigned u = __builtin_bit_cast(unsigned, f);
  u += 0x7FFFu + ((u >> 16) & 1u);   // round-to-nearest-even
  return (unsigned short)(u >> 16);
}

__device__ __forceinline__ float sigmoidf_(float x) {
  return 1.0f / (1.0f + __expf(-x));
}

#define WMMA_BF16(A, B, C) \
  __builtin_amdgcn_wmma_f32_16x16x32_bf16(false, (A), false, (B), (short)0, (C), false, false)

// ---------------------------------------------------------------------------
// Fused GRU cell: gates = xt @ w_ih^T + h @ w_hh^T (+biases), then
// r,z,n nonlinearity and h' = (1-z)*n + z*h.   WMMA bf16 16x16x32.
// Each wave computes a 32(batch) x 16(H-col) output tile: two M-tiles share
// every B (weight) tile -> 6 WMMAs per 10 b128 loads.
// Grid: (Hsz/32, Bsz/32) = (48, 4), block = 64 threads (2 waves).
// ---------------------------------------------------------------------------
__global__ __launch_bounds__(64) void gru_cell_kernel(
    const unsigned short* __restrict__ xt_b, int kx,          // [B, kx] bf16 (padded)
    const unsigned short* __restrict__ wih_b,                 // [3H, kx] bf16
    const unsigned short* __restrict__ whh_b,                 // [3H, H]  bf16
    const float* __restrict__ b_ih, const float* __restrict__ b_hh,
    const float* __restrict__ h_old_f,                        // [B, H] f32
    const unsigned short* __restrict__ h_old_b,               // [B, H] bf16
    float* __restrict__ h_new_f,
    unsigned short* __restrict__ h_new_b)
{
  const int lane   = threadIdx.x & 31;
  const int wave   = threadIdx.x >> 5;
  const int lane15 = lane & 15;
  const int hi     = lane >> 4;
  const int n0 = blockIdx.x * 32 + wave * 16;   // H-column tile base (this wave)
  const int m0 = blockIdx.y * 32;               // batch tile base (2 x 16 rows)

  v8f aIR0{}, aIZ0{}, aIN0{}, aHR0{}, aHZ0{}, aHN0{};
  v8f aIR1{}, aIZ1{}, aIN1{}, aHR1{}, aHZ1{}, aHN1{};

  // ---- hidden-side GEMM: A rows = h_old (two 16-row tiles), B rows = w_hh ----
  {
    const unsigned short* pa0 = h_old_b + (size_t)(m0 + lane15) * Hsz + hi * 8;
    const unsigned short* pa1 = pa0 + (size_t)16 * Hsz;
    const unsigned short* pr = whh_b + (size_t)(0 * Hsz + n0 + lane15) * Hsz + hi * 8;
    const unsigned short* pz = whh_b + (size_t)(1 * Hsz + n0 + lane15) * Hsz + hi * 8;
    const unsigned short* pn = whh_b + (size_t)(2 * Hsz + n0 + lane15) * Hsz + hi * 8;
    for (int k = 0; k < Hsz; k += 32) {
      v16bf a0 = ldAB(pa0 + k);
      v16bf a1 = ldAB(pa1 + k);
      v16bf br = ldAB(pr + k);
      v16bf bz = ldAB(pz + k);
      v16bf bn = ldAB(pn + k);
      aHR0 = WMMA_BF16(a0, br, aHR0);
      aHR1 = WMMA_BF16(a1, br, aHR1);
      aHZ0 = WMMA_BF16(a0, bz, aHZ0);
      aHZ1 = WMMA_BF16(a1, bz, aHZ1);
      aHN0 = WMMA_BF16(a0, bn, aHN0);
      aHN1 = WMMA_BF16(a1, bn, aHN1);
    }
  }
  // ---- input-side GEMM: A rows = xt (padded), B rows = w_ih ----
  {
    const unsigned short* pa0 = xt_b + (size_t)(m0 + lane15) * kx + hi * 8;
    const unsigned short* pa1 = pa0 + (size_t)16 * kx;
    const unsigned short* pr = wih_b + (size_t)(0 * Hsz + n0 + lane15) * kx + hi * 8;
    const unsigned short* pz = wih_b + (size_t)(1 * Hsz + n0 + lane15) * kx + hi * 8;
    const unsigned short* pn = wih_b + (size_t)(2 * Hsz + n0 + lane15) * kx + hi * 8;
    for (int k = 0; k < kx; k += 32) {
      v16bf a0 = ldAB(pa0 + k);
      v16bf a1 = ldAB(pa1 + k);
      v16bf br = ldAB(pr + k);
      v16bf bz = ldAB(pz + k);
      v16bf bn = ldAB(pn + k);
      aIR0 = WMMA_BF16(a0, br, aIR0);
      aIR1 = WMMA_BF16(a1, br, aIR1);
      aIZ0 = WMMA_BF16(a0, bz, aIZ0);
      aIZ1 = WMMA_BF16(a1, bz, aIZ1);
      aIN0 = WMMA_BF16(a0, bn, aIN0);
      aIN1 = WMMA_BF16(a1, bn, aIN1);
    }
  }

  // ---- epilogue: C/D layout -> element i of lane = (mbase + i + 8*hi, n0 + lane15)
  const int col = n0 + lane15;
  const float bir = b_ih[col], biz = b_ih[Hsz + col], bin = b_ih[2 * Hsz + col];
  const float bhr = b_hh[col], bhz = b_hh[Hsz + col], bhn = b_hh[2 * Hsz + col];

#pragma unroll
  for (int mt = 0; mt < 2; ++mt) {
    const int mbase = m0 + mt * 16;
    const v8f& ir = mt ? aIR1 : aIR0;
    const v8f& iz = mt ? aIZ1 : aIZ0;
    const v8f& in_ = mt ? aIN1 : aIN0;
    const v8f& hr = mt ? aHR1 : aHR0;
    const v8f& hz = mt ? aHZ1 : aHZ0;
    const v8f& hn = mt ? aHN1 : aHN0;
#pragma unroll
    for (int i = 0; i < 8; ++i) {
      const int row = mbase + i + 8 * hi;
      const size_t idx = (size_t)row * Hsz + col;
      float r = sigmoidf_(ir[i] + bir + hr[i] + bhr);
      float z = sigmoidf_(iz[i] + biz + hz[i] + bhz);
      float n = tanhf(in_[i] + bin + r * (hn[i] + bhn));
      float hp = h_old_f[idx];
      float hn2 = (1.0f - z) * n + z * hp;
      h_new_f[idx] = hn2;
      h_new_b[idx] = f2bf(hn2);
    }
  }
}

// ---------------------------------------------------------------------------
// fc0 + LayerNorm + residual feedback (tiny: 128x66x1536 fp32)
// one block per batch row, 96 threads (threads 66..95 only zero-pad dec_b)
// ---------------------------------------------------------------------------
__global__ __launch_bounds__(96) void fc_ln_kernel(
    const float* __restrict__ h1_f, const float* __restrict__ fc_w,
    const float* __restrict__ fc_b, const float* __restrict__ ln_a,
    const float* __restrict__ ln_b, const float* __restrict__ dec_old_f,
    float* __restrict__ out, int t,
    float* __restrict__ dec_new_f, unsigned short* __restrict__ dec_new_b)
{
  __shared__ float sd[Csz];
  const int b = blockIdx.x, c = threadIdx.x;
  if (c < Csz) {
    const float* w = fc_w + (size_t)c * Hsz;
    const float* h = h1_f + (size_t)b * Hsz;
    float s = fc_b[c];
    for (int k = 0; k < Hsz; ++k) s = fmaf(w[k], h[k], s);
    sd[c] = s;
  }
  __syncthreads();
  if (c < Csz) {
    float mu = 0.f;
    for (int j = 0; j < Csz; ++j) mu += sd[j];
    mu *= (1.0f / Csz);
    float var = 0.f;
    for (int j = 0; j < Csz; ++j) { float d = sd[j] - mu; var += d * d; }
    var *= (1.0f / Csz);
    float y = (sd[c] - mu) * rsqrtf(var + EPSV) * ln_a[c] + ln_b[c] + dec_old_f[b * Csz + c];
    out[(size_t)b * Tsz * Csz + (size_t)t * Csz + c] = y;
    dec_new_f[b * Csz + c] = y;
    dec_new_b[b * KX0 + c] = f2bf(y);
  } else {
    dec_new_b[b * KX0 + c] = 0;   // zero-pad cols 66..95
  }
}

// ---------------------------------------------------------------------------
// helpers: fp32 -> bf16 (with column padding), x packing, dec init, zeroing
// ---------------------------------------------------------------------------
__global__ void convert_pad_kernel(const float* __restrict__ src,
                                   unsigned short* __restrict__ dst,
                                   int rows, int sc, int dc) {
  int idx = blockIdx.x * blockDim.x + threadIdx.x;
  if (idx >= rows * dc) return;
  int r = idx / dc, c = idx - r * dc;
  float v = (c < sc) ? src[(size_t)r * sc + c] : 0.0f;
  dst[idx] = f2bf(v);
}

__global__ void pack_x_kernel(const float* __restrict__ x,
                              unsigned short* __restrict__ xb) {
  int idx = blockIdx.x * blockDim.x + threadIdx.x;
  if (idx >= Tsz * Bsz * KX0) return;
  int t = idx / (Bsz * KX0);
  int rem = idx - t * (Bsz * KX0);
  int b = rem / KX0, c = rem - b * KX0;
  float v = (c < Csz) ? x[(size_t)b * Tsz * Csz + (size_t)t * Csz + c] : 0.0f;
  xb[idx] = f2bf(v);
}

__global__ void init_dec_kernel(const float* __restrict__ x,
                                float* __restrict__ dec_f,
                                unsigned short* __restrict__ dec_b) {
  int idx = blockIdx.x * blockDim.x + threadIdx.x;
  if (idx >= Bsz * KX0) return;
  int b = idx / KX0, c = idx - b * KX0;
  if (c < Csz) {
    float v = x[(size_t)b * Tsz * Csz + (size_t)(Tsz - 1) * Csz + c];
    dec_f[b * Csz + c] = v;
    dec_b[idx] = f2bf(v);
  } else {
    dec_b[idx] = 0;
  }
}

__global__ void zero_kernel(uint4* __restrict__ p, int n16) {
  int i = blockIdx.x * blockDim.x + threadIdx.x;
  if (i < n16) p[i] = make_uint4(0u, 0u, 0u, 0u);
}

// ---------------------------------------------------------------------------
extern "C" void kernel_launch(void* const* d_in, const int* in_sizes, int n_in,
                              void* d_out, int out_size, void* d_ws, size_t ws_size,
                              hipStream_t stream) {
  (void)in_sizes; (void)n_in; (void)out_size; (void)ws_size;
  const float* x     = (const float*)d_in[0];
  const float* w_ih0 = (const float*)d_in[1];
  const float* w_hh0 = (const float*)d_in[2];
  const float* b_ih0 = (const float*)d_in[3];
  const float* b_hh0 = (const float*)d_in[4];
  const float* w_ih1 = (const float*)d_in[5];
  const float* w_hh1 = (const float*)d_in[6];
  const float* b_ih1 = (const float*)d_in[7];
  const float* b_hh1 = (const float*)d_in[8];
  const float* fc_w  = (const float*)d_in[9];
  const float* fc_b  = (const float*)d_in[10];
  const float* ln_a  = (const float*)d_in[11];
  const float* ln_b  = (const float*)d_in[12];
  float* out = (float*)d_out;

  char* ws = (char*)d_ws;
  size_t off = 0;
  auto take = [&](size_t bytes) -> char* {
    char* p = ws + off;
    off += (bytes + 255) & ~(size_t)255;
    return p;
  };

  // --- hidden-state region first (zeroed in one shot) ---
  float*          h0f[2]; unsigned short* h0b[2];
  float*          h1f[2]; unsigned short* h1b[2];
  h0f[0] = (float*)take((size_t)Bsz * Hsz * 4);
  h0f[1] = (float*)take((size_t)Bsz * Hsz * 4);
  h1f[0] = (float*)take((size_t)Bsz * Hsz * 4);
  h1f[1] = (float*)take((size_t)Bsz * Hsz * 4);
  h0b[0] = (unsigned short*)take((size_t)Bsz * Hsz * 2);
  h0b[1] = (unsigned short*)take((size_t)Bsz * Hsz * 2);
  h1b[0] = (unsigned short*)take((size_t)Bsz * Hsz * 2);
  h1b[1] = (unsigned short*)take((size_t)Bsz * Hsz * 2);
  const size_t hbytes = off;   // all sizes are multiples of 256 -> dense

  unsigned short* wih0b = (unsigned short*)take((size_t)3 * Hsz * KX0 * 2);
  unsigned short* whh0b = (unsigned short*)take((size_t)3 * Hsz * Hsz * 2);
  unsigned short* wih1b = (unsigned short*)take((size_t)3 * Hsz * Hsz * 2);
  unsigned short* whh1b = (unsigned short*)take((size_t)3 * Hsz * Hsz * 2);
  unsigned short* xb    = (unsigned short*)take((size_t)Tsz * Bsz * KX0 * 2);
  float* decf[2]; unsigned short* decb[2];
  decf[0] = (float*)take((size_t)Bsz * Csz * 4);
  decf[1] = (float*)take((size_t)Bsz * Csz * 4);
  decb[0] = (unsigned short*)take((size_t)Bsz * KX0 * 2);
  decb[1] = (unsigned short*)take((size_t)Bsz * KX0 * 2);

  // --- one-time per-launch prep (deterministic, recomputed every call) ---
  {
    int n16 = (int)(hbytes / 16);
    zero_kernel<<<(n16 + 255) / 256, 256, 0, stream>>>((uint4*)ws, n16);

    int n;
    n = 3 * Hsz * KX0;
    convert_pad_kernel<<<(n + 255) / 256, 256, 0, stream>>>(w_ih0, wih0b, 3 * Hsz, Csz, KX0);
    n = 3 * Hsz * Hsz;
    convert_pad_kernel<<<(n + 255) / 256, 256, 0, stream>>>(w_hh0, whh0b, 3 * Hsz, Hsz, Hsz);
    convert_pad_kernel<<<(n + 255) / 256, 256, 0, stream>>>(w_ih1, wih1b, 3 * Hsz, Hsz, Hsz);
    convert_pad_kernel<<<(n + 255) / 256, 256, 0, stream>>>(w_hh1, whh1b, 3 * Hsz, Hsz, Hsz);
    n = Tsz * Bsz * KX0;
    pack_x_kernel<<<(n + 255) / 256, 256, 0, stream>>>(x, xb);
    n = Bsz * KX0;
    init_dec_kernel<<<(n + 255) / 256, 256, 0, stream>>>(x, decf[0], decb[0]);
  }

  const dim3 gc(Hsz / 32, Bsz / 32);   // (48, 4), 64-thread blocks (2 waves)
  int c0 = 0, c1 = 0, cd = 0;

  // --- encoder: first T-1 frames ---
  for (int t = 0; t < Tsz - 1; ++t) {
    gru_cell_kernel<<<gc, 64, 0, stream>>>(
        xb + (size_t)t * Bsz * KX0, KX0, wih0b, whh0b, b_ih0, b_hh0,
        h0f[c0], h0b[c0], h0f[1 - c0], h0b[1 - c0]);
    c0 ^= 1;
    gru_cell_kernel<<<gc, 64, 0, stream>>>(
        h0b[c0], Hsz, wih1b, whh1b, b_ih1, b_hh1,
        h1f[c1], h1b[c1], h1f[1 - c1], h1b[1 - c1]);
    c1 ^= 1;
  }

  // --- decoder: T steps with fc + LN + residual feedback ---
  for (int t = 0; t < Tsz; ++t) {
    gru_cell_kernel<<<gc, 64, 0, stream>>>(
        decb[cd], KX0, wih0b, whh0b, b_ih0, b_hh0,
        h0f[c0], h0b[c0], h0f[1 - c0], h0b[1 - c0]);
    c0 ^= 1;
    gru_cell_kernel<<<gc, 64, 0, stream>>>(
        h0b[c0], Hsz, wih1b, whh1b, b_ih1, b_hh1,
        h1f[c1], h1b[c1], h1f[1 - c1], h1b[1 - c1]);
    c1 ^= 1;
    fc_ln_kernel<<<Bsz, 96, 0, stream>>>(
        h1f[c1], fc_w, fc_b, ln_a, ln_b, decf[cd],
        out, t, decf[1 - cd], decb[1 - cd]);
    cd ^= 1;
  }
}
